// ModuleSelector_21053929685471
// MI455X (gfx1250) — matrix operand order repaired and checked by reading, not compiled, not run on hardware
//
#include <hip/hip_runtime.h>

#define NFEATS   131072
#define INSZ     256
#define HID      1024
#define OUTSZ    256
#define NEXP     16
#define TM       64                        // rows per block (four 16-row WMMA tiles)
#define HSTRIDE  1032                      // 1024 + 8 bf16 pad -> row shift of 4 LDS banks
#define MAXTILES (NFEATS/TM + NEXP)        // 2064 upper bound on per-expert 64-row tiles

typedef __attribute__((ext_vector_type(16))) __bf16 v16bf;
typedef __attribute__((ext_vector_type(8)))  __bf16 v8bf;
typedef __attribute__((ext_vector_type(8)))  float  v8f;

// ---------------- routing / bookkeeping kernels ----------------

__global__ void init_counts_kernel(int* counts) {
    if (threadIdx.x < NEXP) counts[threadIdx.x] = 0;
}

__global__ void count_ids_kernel(const int* __restrict__ ids, int* __restrict__ counts) {
    int i = blockIdx.x * blockDim.x + threadIdx.x;
    atomicAdd(&counts[ids[i] & (NEXP - 1)], 1);
}

__global__ void scan16_kernel(const int* __restrict__ counts, int* __restrict__ offsets,
                              int* __restrict__ cursor, int* __restrict__ tileBase) {
    if (threadIdx.x == 0) {
        int o = 0, t = 0;
        for (int e = 0; e < NEXP; ++e) {
            offsets[e] = o; cursor[e] = o; tileBase[e] = t;
            t += (counts[e] + TM - 1) / TM;
            o += counts[e];
        }
        offsets[NEXP] = o; tileBase[NEXP] = t;
    }
}

__global__ void scatter_ids_kernel(const int* __restrict__ ids, int* __restrict__ cursor,
                                   int* __restrict__ perm) {
    int i = blockIdx.x * blockDim.x + threadIdx.x;
    int e = ids[i] & (NEXP - 1);
    int pos = atomicAdd(&cursor[e], 1);
    perm[pos] = i;
}

// ---------------- fp32 -> bf16 transposing weight pack ----------------
// in : per-expert [R][C] fp32 row-major   (W as stored: [K][N])
// out: per-expert [C][R] bf16 row-major   (W^T: [N][K], contiguous K for B-frags)
__global__ void transpose_cvt_kernel(const float* __restrict__ in, __bf16* __restrict__ out,
                                     int R, int C) {
    __shared__ float tile[32][33];
    const int e = blockIdx.z;
    in  += (size_t)e * R * C;
    out += (size_t)e * R * C;
    const int c0 = blockIdx.x * 32, r0 = blockIdx.y * 32;
    for (int i = threadIdx.y; i < 32; i += 8)
        tile[i][threadIdx.x] = in[(size_t)(r0 + i) * C + (c0 + threadIdx.x)];
    __syncthreads();
    for (int i = threadIdx.y; i < 32; i += 8)
        out[(size_t)(c0 + i) * R + (r0 + threadIdx.x)] = (__bf16)tile[threadIdx.x][i];
}

// ---------------- fragment loaders (ISA 7.12.2 layouts, wave32) ----------------

static __device__ inline v16bf cat8(v8bf lo, v8bf hi) {
    return __builtin_shufflevector(lo, hi, 0,1,2,3,4,5,6,7,8,9,10,11,12,13,14,15);
}
// B 32x16 bf16: lane n = lane&15, K octets at (k0+half*8) and (k0+16+half*8)
static __device__ inline v16bf load_b_frag(const __bf16* p) {
    return cat8(*(const v8bf*)p, *(const v8bf*)(p + 16));
}
// A 16x32 bf16 from LDS: K octets at (k0+half*8) and (k0+16+half*8)
static __device__ inline v16bf load_a_lds(const __bf16* p) {
    return cat8(*(const v8bf*)p, *(const v8bf*)(p + 16));
}
// A 16x32 built on the fly from fp32 x row (p points at k0 + half*8)
static __device__ inline v16bf load_a_f32(const float* p) {
    float4 c0 = ((const float4*)p)[0];
    float4 c1 = ((const float4*)p)[1];
    float4 c2 = ((const float4*)(p + 16))[0];
    float4 c3 = ((const float4*)(p + 16))[1];
    v16bf r;
    r[0]=(__bf16)c0.x; r[1]=(__bf16)c0.y; r[2]=(__bf16)c0.z; r[3]=(__bf16)c0.w;
    r[4]=(__bf16)c1.x; r[5]=(__bf16)c1.y; r[6]=(__bf16)c1.z; r[7]=(__bf16)c1.w;
    r[8]=(__bf16)c2.x; r[9]=(__bf16)c2.y; r[10]=(__bf16)c2.z; r[11]=(__bf16)c2.w;
    r[12]=(__bf16)c3.x; r[13]=(__bf16)c3.y; r[14]=(__bf16)c3.z; r[15]=(__bf16)c3.w;
    return r;
}

// ---------------- fused per-tile 2-layer expert MLP ----------------
// Block = 512 threads = 16 wave32. One 64-row tile of one expert per block.
// Every weight fragment is loaded exactly ONCE per block in both layers
// (waves partition n; each wave covers all 4 row-tiles -> 4x A/B register reuse).
// L1: wave w -> hidden cols [w*64, w*64+64) x 4 row-tiles, K=256.
//     Per k-step: 4 A frags + 4 B frags (clauses) -> 16 back-to-back WMMAs.
// L2: wave w -> out cols [w*16, w*16+16) x 4 row-tiles, K=1024.
//     Per k-step: 4 LDS A frags + 1 B frag -> 4 WMMAs. Scatter rows via perm.
__global__ __launch_bounds__(512) void moe_mlp_kernel(
    const float* __restrict__ x,  const float* __restrict__ b1,
    const float* __restrict__ b2, const __bf16* __restrict__ W1T,
    const __bf16* __restrict__ W2T, const int* __restrict__ perm,
    const int* __restrict__ offsets, const int* __restrict__ tileBase,
    float* __restrict__ out)
{
    __shared__ __attribute__((aligned(16))) __bf16 hbuf[TM][HSTRIDE];
    __shared__ int rowsafe[TM];
    __shared__ int roworig[TM];

    const int b = blockIdx.x;
    if (b >= tileBase[NEXP]) return;

    int e = 0;
    for (int i = 1; i < NEXP; ++i) if (b >= tileBase[i]) e = i;   // uniform scalar search
    const int tile     = b - tileBase[e];
    const int rowStart = offsets[e] + tile * TM;
    const int rowEnd   = offsets[e + 1];

    if (threadIdx.x < TM) {
        int idx  = rowStart + (int)threadIdx.x;
        int orig = (idx < rowEnd) ? perm[idx] : -1;
        roworig[threadIdx.x] = orig;
        rowsafe[threadIdx.x] = (orig >= 0) ? orig : perm[rowStart]; // EXEC stays all-1 for WMMA
    }
    __syncthreads();

    const int lane = threadIdx.x & 31;
    const int wave = threadIdx.x >> 5;     // 0..15
    const int col  = lane & 15;            // A row m within tile / B,C,D col n within tile
    const int half = lane >> 4;

    // ---------- layer 1: h = relu(x @ W1 + b1) ----------
    const float* xrow[4];
#pragma unroll
    for (int rt = 0; rt < 4; ++rt)
        xrow[rt] = x + (size_t)rowsafe[rt * 16 + col] * INSZ;
    const __bf16* w1e = W1T + (size_t)e * HID * INSZ;

    v8f acc1[16] = {};                      // acc1[rt*4 + t]
    for (int kk = 0; kk < INSZ / 32; ++kk) {
        const int k0 = kk * 32;
        v16bf a[4];
#pragma unroll
        for (int rt = 0; rt < 4; ++rt)
            a[rt] = load_a_f32(xrow[rt] + k0 + half * 8);
        v16bf bm[4];
#pragma unroll
        for (int t = 0; t < 4; ++t) {
            const int n = wave * 64 + t * 16 + col;
            bm[t] = load_b_frag(w1e + (size_t)n * INSZ + k0 + half * 8);
        }
#pragma unroll
        for (int t = 0; t < 4; ++t)
#pragma unroll
            for (int rt = 0; rt < 4; ++rt)
                acc1[rt * 4 + t] = __builtin_amdgcn_wmma_f32_16x16x32_bf16(
                    false, a[rt], false, bm[t], (short)0, acc1[rt * 4 + t], false, false);
    }
#pragma unroll
    for (int t = 0; t < 4; ++t) {
        const int n = wave * 64 + t * 16 + col;
        const float bias = b1[e * HID + n];
#pragma unroll
        for (int rt = 0; rt < 4; ++rt)
#pragma unroll
            for (int r = 0; r < 8; ++r) {
                float v = acc1[rt * 4 + t][r] + bias;
                v = v > 0.0f ? v : 0.0f;
                hbuf[rt * 16 + r + half * 8][n] = (__bf16)v;
            }
    }
    __syncthreads();

    // ---------- layer 2: y = h @ W2 + b2 ----------
    const __bf16* hrow[4];
#pragma unroll
    for (int rt = 0; rt < 4; ++rt)
        hrow[rt] = &hbuf[rt * 16 + col][0];
    const int     n2    = wave * 16 + col;
    const __bf16* w2col = W2T + ((size_t)e * OUTSZ + n2) * HID;

    v8f acc2[4] = {};
    for (int kk = 0; kk < HID / 32; ++kk) {
        const int k0 = kk * 32;
        v16bf bm = load_b_frag(w2col + k0 + half * 8);
        v16bf a[4];
#pragma unroll
        for (int rt = 0; rt < 4; ++rt)
            a[rt] = load_a_lds(hrow[rt] + k0 + half * 8);
#pragma unroll
        for (int rt = 0; rt < 4; ++rt)
            acc2[rt] = __builtin_amdgcn_wmma_f32_16x16x32_bf16(
                false, a[rt], false, bm, (short)0, acc2[rt], false, false);
    }
    {
        const float bias = b2[e * OUTSZ + n2];
#pragma unroll
        for (int rt = 0; rt < 4; ++rt)
#pragma unroll
            for (int r = 0; r < 8; ++r) {
                const int orig = roworig[rt * 16 + r + half * 8];
                if (orig >= 0) out[(size_t)orig * OUTSZ + n2] = acc2[rt][r] + bias;
            }
    }
}

// ---------------- launch ----------------

extern "C" void kernel_launch(void* const* d_in, const int* in_sizes, int n_in,
                              void* d_out, int out_size, void* d_ws, size_t ws_size,
                              hipStream_t stream) {
    (void)in_sizes; (void)n_in; (void)out_size; (void)ws_size;
    const float* x   = (const float*)d_in[0];
    const float* W1  = (const float*)d_in[1];
    const float* b1  = (const float*)d_in[2];
    const float* W2  = (const float*)d_in[3];
    const float* b2  = (const float*)d_in[4];
    const int*   ids = (const int*)d_in[5];
    float* out = (float*)d_out;

    char* ws = (char*)d_ws;                       // workspace layout (bytes)
    int* counts   = (int*)(ws + 0);               // 16
    int* cursor   = (int*)(ws + 64);              // 16
    int* offsets  = (int*)(ws + 128);             // 17
    int* tileBase = (int*)(ws + 256);             // 17
    int* perm     = (int*)(ws + 512);             // 131072 ints = 512 KiB
    __bf16* W1T   = (__bf16*)(ws + 524800);                 // 16*1024*256*2 = 8 MiB
    __bf16* W2T   = (__bf16*)(ws + 524800 + 8388608);       // 16*256*1024*2 = 8 MiB

    init_counts_kernel<<<1, 32, 0, stream>>>(counts);
    count_ids_kernel<<<NFEATS / 256, 256, 0, stream>>>(ids, counts);
    scan16_kernel<<<1, 32, 0, stream>>>(counts, offsets, cursor, tileBase);
    scatter_ids_kernel<<<NFEATS / 256, 256, 0, stream>>>(ids, cursor, perm);

    dim3 tb(32, 8);
    transpose_cvt_kernel<<<dim3(HID / 32, INSZ / 32, NEXP), tb, 0, stream>>>(W1, W1T, INSZ, HID);
    transpose_cvt_kernel<<<dim3(OUTSZ / 32, HID / 32, NEXP), tb, 0, stream>>>(W2, W2T, HID, OUTSZ);

    moe_mlp_kernel<<<MAXTILES, 512, 0, stream>>>(x, b1, b2, W1T, W2T,
                                                 perm, offsets, tileBase, out);
}
